// Model_13718125543908
// MI455X (gfx1250) — matrix operand order repaired and checked by reading, not compiled, not run on hardware
//
#include <hip/hip_runtime.h>
#include <stdint.h>

#define D_  1024
#define T_  1024
#define NB  32
#define NC  20
#define TP2 1026   // T + 2 (pad 1 each side for k=3 conv)

typedef __attribute__((ext_vector_type(16))) _Float16 v16h;
typedef __attribute__((ext_vector_type(8)))  float    v8f;

union Frag16 { uint4 q[2]; v16h v; };

// ---------------- prep: x -> padded fp16 (stream, n, t, d)  [coalesced copy] ----------------
__global__ void k_prep_x(const float* __restrict__ x, _Float16* __restrict__ Xp, long total) {
    long id = (long)blockIdx.x * blockDim.x + threadIdx.x;
    if (id >= total) return;
    int d = (int)(id % D_);
    long rest = id / D_;
    int tt = (int)(rest % TP2); rest /= TP2;
    int n = (int)(rest % NB);
    int s = (int)(rest / NB);
    float v = 0.f;
    if (tt >= 1 && tt <= T_)
        v = x[(((long)n * T_ + (tt - 1)) * 2 + s) * D_ + d];
    Xp[id] = (_Float16)v;
}

// ---------------- prep: w1 (D,D,3) -> fp16 tap-major (e,tap,o,i) ----------------
__global__ void k_prep_w(const float* w0, const float* w1, const float* w2, const float* w3,
                         _Float16* __restrict__ Wt) {
    int id = blockIdx.x * blockDim.x + threadIdx.x;
    if (id >= 4 * 3 * D_ * D_) return;
    int i   = id % D_;
    int o   = (id / D_) % D_;
    int tap = (id / (D_ * D_)) % 3;
    int e   = id / (3 * D_ * D_);
    const float* w = (e == 0) ? w0 : (e == 1) ? w1 : (e == 2) ? w2 : w3;
    Wt[id] = (_Float16)w[(o * D_ + i) * 3 + tap];
}

// ---------------- prep: stacked second-layer weights (32 x 4096) ----------------
__global__ void k_prep_w2(const float* crw2, const float* cfw2, const float* arw2, const float* afw2,
                          _Float16* __restrict__ W2A) {
    int id = blockIdx.x * blockDim.x + threadIdx.x;
    if (id >= 32 * 4096) return;
    int kk = id % 4096;
    int m  = id / 4096;
    float v = 0.f;
    if (m < NC) {
        if (kk < 1024)       v = crw2[m * D_ + kk];
        else if (kk < 2048)  v = cfw2[m * D_ + (kk - 1024)];
    } else if (m == NC) {
        if (kk >= 2048 && kk < 3072) v = arw2[kk - 2048];
        else if (kk >= 3072)         v = afw2[kk - 3072];
    }
    W2A[id] = (_Float16)v;
}

// ---------------- conv1 (k=3, pad 1) as 3-tap WMMA GEMM + bias + relu ----------------
// block: 256 thr = 8 waves; tile 128(M=d) x 128(N=t); K = D per tap.
// LDS: A tiles [tap][m][k] (row=K contiguous); B tile [t][k] (K contiguous per t).
// All WMMA fragments load as 2 x ds_load_b128.
__global__ __launch_bounds__(256) void k_conv_gemm(
    const _Float16* __restrict__ Wt, const _Float16* __restrict__ Xp,
    const float* b1_0, const float* b1_1, const float* b1_2, const float* b1_3,
    _Float16* __restrict__ H) {
    __shared__ unsigned short As[3][128][40];   // 3 taps, 128 M rows, 32 K (+pad to 80B stride)
    __shared__ unsigned short Bs[132][40];      // 132 t rows (128 + 2 halo + pad), 32 K

    const int e    = blockIdx.z;
    const int n    = blockIdx.y;
    const int m0   = (blockIdx.x >> 3) * 128;
    const int t0   = (blockIdx.x & 7) * 128;
    const int tid  = threadIdx.x;
    const int lane = tid & 31;
    const int wave = tid >> 5;
    const int wm   = (wave >> 1) * 32;   // 4 wave-rows of 32
    const int wn   = (wave & 1) * 64;    // 2 wave-cols of 64
    const int s    = e & 1;              // cr,ar -> rgb(0); cf,af -> flow(1)
    const float* b1 = (e == 0) ? b1_0 : (e == 1) ? b1_1 : (e == 2) ? b1_2 : b1_3;

    const _Float16* Wbase = Wt + (size_t)e * 3 * D_ * D_;
    const _Float16* Xbase = Xp + ((size_t)(s * NB + n)) * TP2 * D_;

    v8f acc[2][4];
    const v8f vz = {0.f, 0.f, 0.f, 0.f, 0.f, 0.f, 0.f, 0.f};
    for (int i = 0; i < 2; ++i)
        for (int j = 0; j < 4; ++j) acc[i][j] = vz;

    const int lr = tid >> 1;           // row this thread loads (A: m, B: t_local)
    const int lk = (tid & 1) * 16;     // k-offset (halfs)

    for (int k0 = 0; k0 < D_; k0 += 32) {
        // A tiles: one per tap, 128x32, b128 loads/stores
        for (int tap = 0; tap < 3; ++tap) {
            const _Float16* ga = Wbase + ((size_t)tap * D_ + (m0 + lr)) * D_ + k0 + lk;
            uint4 q0 = *(const uint4*)ga;
            uint4 q1 = *(const uint4*)(ga + 8);
            *(uint4*)&As[tap][lr][lk]     = q0;
            *(uint4*)&As[tap][lr][lk + 8] = q1;
        }
        // B tile: rows t0..t0+127 always in-bounds (t0 <= 896, TP2 = 1026)
        {
            const _Float16* gb = Xbase + (size_t)(t0 + lr) * D_ + k0 + lk;
            uint4 q0 = *(const uint4*)gb;
            uint4 q1 = *(const uint4*)(gb + 8);
            *(uint4*)&Bs[lr][lk]     = q0;
            *(uint4*)&Bs[lr][lk + 8] = q1;
        }
        // halo rows 128..131 (guarded, zero-fill past padded end)
        if (tid < 8) {
            int tl = 128 + (tid >> 1);
            int kf = (tid & 1) * 16;
            uint4 q0 = {0u, 0u, 0u, 0u}, q1 = {0u, 0u, 0u, 0u};
            if (t0 + tl < TP2) {
                const _Float16* gb = Xbase + (size_t)(t0 + tl) * D_ + k0 + kf;
                q0 = *(const uint4*)gb;
                q1 = *(const uint4*)(gb + 8);
            }
            *(uint4*)&Bs[tl][kf]     = q0;
            *(uint4*)&Bs[tl][kf + 8] = q1;
        }
        __syncthreads();

        const int arow = lane & 15;
        const int kh   = (lane >> 4) * 8;    // A: elements 0-7 = K kh.., 8-15 = K 16+kh..
        const int kb   = (lane >> 4) * 16;   // B: same K order as A (kh.., 16+kh..)

        for (int tap = 0; tap < 3; ++tap) {
            Frag16 fa[2];
#pragma unroll
            for (int mi = 0; mi < 2; ++mi) {
                int r = wm + mi * 16 + arow;
                fa[mi].q[0] = *(const uint4*)&As[tap][r][kh];
                fa[mi].q[1] = *(const uint4*)&As[tap][r][16 + kh];
            }
            Frag16 fb[4];
#pragma unroll
            for (int ni = 0; ni < 4; ++ni) {
                int col = wn + ni * 16 + arow + tap;
                fb[ni].q[0] = *(const uint4*)&Bs[col][kh];
                fb[ni].q[1] = *(const uint4*)&Bs[col][16 + kh];
            }
#pragma unroll
            for (int mi = 0; mi < 2; ++mi)
#pragma unroll
                for (int ni = 0; ni < 4; ++ni)
                    acc[mi][ni] = __builtin_amdgcn_wmma_f32_16x16x32_f16(
                        false, fa[mi].v, false, fb[ni].v, (short)0, acc[mi][ni], false, false);
        }
        __syncthreads();
    }

    // epilogue: bias + relu -> fp16, layout H[((n*T+t)*4 + e)*D + d], b128 stores
    for (int mi = 0; mi < 2; ++mi) {
        int dbase = m0 + wm + mi * 16 + (lane >> 4) * 8;
        for (int ni = 0; ni < 4; ++ni) {
            int t = t0 + wn + ni * 16 + (lane & 15);
            union { _Float16 h[8]; uint4 q; } pk;
#pragma unroll
            for (int i = 0; i < 8; ++i) {
                float v = acc[mi][ni][i] + b1[dbase + i];
                pk.h[i] = (_Float16)(v > 0.f ? v : 0.f);
            }
            *(uint4*)&H[(((size_t)n * T_ + t) * 4 + e) * D_ + dbase] = pk.q;
        }
    }
}

// ---------------- second layer: logits(21 x T) = W2A(32x4096) @ Hcat ----------------
__global__ __launch_bounds__(256) void k_l2_gemm(
    const _Float16* __restrict__ W2A, const _Float16* __restrict__ H,
    const float* crb2, const float* cfb2, const float* arb2, const float* afb2,
    float* __restrict__ logits) {
    __shared__ unsigned short As2[32][40];
    __shared__ unsigned short Bs2[128][40];

    const int n    = blockIdx.y;
    const int t0   = blockIdx.x * 128;
    const int tid  = threadIdx.x;
    const int lane = tid & 31;
    const int wave = tid >> 5;

    v8f acc[2];
    const v8f vz = {0.f, 0.f, 0.f, 0.f, 0.f, 0.f, 0.f, 0.f};
    acc[0] = vz; acc[1] = vz;

    const int arr = tid >> 3;          // A row, 4 halfs/thread
    const int ark = (tid & 7) * 4;
    const int brc = tid >> 1;          // B col (t), 16 halfs/thread contiguous in K
    const int brk = (tid & 1) * 16;

    for (int k0 = 0; k0 < 4096; k0 += 32) {
        *(uint2*)&As2[arr][ark] = *(const uint2*)(W2A + (size_t)arr * 4096 + k0 + ark);
        const _Float16* gb = H + (((size_t)n * T_ + t0 + brc) * 4096) + k0 + brk;
        uint4 q0 = *(const uint4*)gb;
        uint4 q1 = *(const uint4*)(gb + 8);
        *(uint4*)&Bs2[brc][brk]     = q0;
        *(uint4*)&Bs2[brc][brk + 8] = q1;
        __syncthreads();

        const int arow = lane & 15;
        const int kh   = (lane >> 4) * 8;
        const int kb   = (lane >> 4) * 16;

        Frag16 fa[2], fb;
#pragma unroll
        for (int mi = 0; mi < 2; ++mi) {
            int r = mi * 16 + arow;
            fa[mi].q[0] = *(const uint4*)&As2[r][kh];
            fa[mi].q[1] = *(const uint4*)&As2[r][16 + kh];
        }
        const int col = wave * 16 + arow;
        fb.q[0] = *(const uint4*)&Bs2[col][kh];
        fb.q[1] = *(const uint4*)&Bs2[col][16 + kh];

#pragma unroll
        for (int mi = 0; mi < 2; ++mi)
            acc[mi] = __builtin_amdgcn_wmma_f32_16x16x32_f16(
                false, fa[mi].v, false, fb.v, (short)0, acc[mi], false, false);
        __syncthreads();
    }

    int t = t0 + wave * 16 + (lane & 15);
    for (int mi = 0; mi < 2; ++mi) {
        int mb = mi * 16 + (lane >> 4) * 8;
#pragma unroll
        for (int i = 0; i < 8; ++i) {
            int m = mb + i;
            if (m <= NC) {
                float bias = (m < NC) ? (crb2[m] + cfb2[m]) : (arb2[0] + afb2[0]);
                logits[((size_t)n * T_ + t) * 21 + m] = acc[mi][i] + bias;
            }
        }
    }
}

// ---------------- softmax/sigmoid -> aas, seg_score ----------------
__global__ void k_softmax(const float* __restrict__ logits,
                          float* __restrict__ out_aas, float* __restrict__ out_seg) {
    int id = blockIdx.x * blockDim.x + threadIdx.x;
    if (id >= NB * T_) return;
    const float* l = logits + (size_t)id * 21;
    float v[NC];
    float mx = -3.4e38f;
    for (int c = 0; c < NC; ++c) { v[c] = l[c]; mx = fmaxf(mx, v[c]); }
    float ssum = 0.f;
    for (int c = 0; c < NC; ++c) { v[c] = __expf(v[c] - mx); ssum += v[c]; }
    float inv = 1.f / ssum;
    float a = 1.f / (1.f + __expf(-l[NC]));
    out_aas[id] = a;
    for (int c = 0; c < NC; ++c)
        out_seg[(size_t)id * NC + c] = (v[c] * inv + a) * 0.5f;
}

// ---------------- temporal clustering: stable desc sort + serial scan ----------------
__global__ __launch_bounds__(512) void k_cluster(const float* __restrict__ seg,
                                                 float* __restrict__ mask0) {
    __shared__ uint64_t keys[1024];
    __shared__ float asn[1024];
    const int r = blockIdx.x;       // n*NC + c
    const int n = r / NC, c = r % NC;
    const int tid = threadIdx.x;
    for (int t = tid; t < T_; t += 512) {
        float f = seg[((size_t)n * T_ + t) * NC + c];
        uint32_t u = __float_as_uint(f);
        uint32_t m = (u & 0x80000000u) ? ~u : (u | 0x80000000u);   // monotone asc
        keys[t] = ((uint64_t)(~m) << 32) | (uint32_t)t;            // asc key = desc value, stable
    }
    __syncthreads();
    for (unsigned k = 2; k <= 1024; k <<= 1) {
        for (unsigned j = k >> 1; j > 0; j >>= 1) {
            for (unsigned i = tid; i < 1024; i += 512) {
                unsigned ixj = i ^ j;
                if (ixj > i) {
                    uint64_t a = keys[i], b = keys[ixj];
                    bool up = ((i & k) == 0);
                    if (up ? (a > b) : (a < b)) { keys[i] = b; keys[ixj] = a; }
                }
            }
            __syncthreads();
        }
    }
    if (tid == 0) {
        auto dec = [](uint64_t key) -> float {
            uint32_t m = ~(uint32_t)(key >> 32);
            uint32_t u = (m & 0x80000000u) ? (m & 0x7fffffffu) : ~m;
            return __uint_as_float(u);
        };
        float ps = dec(keys[0]),     pn = 1.f;
        float ns = dec(keys[T_ - 1]), nn = 1.f;
        asn[0] = 1.f;
        for (int jj = 1; jj <= T_ - 2; ++jj) {
            float v = dec(keys[jj]);
            float w = 1.f / (float)(jj + 1);
            float cf = (fabsf(v - ps / pn) <= fabsf(v - ns / nn)) ? 1.f : 0.f;
            ps += cf * v * w;       pn += cf * w;
            ns += (1.f - cf) * v;   nn += (1.f - cf);
            asn[jj] = cf;
        }
    }
    __syncthreads();
    for (int i = tid; i < T_; i += 512) {
        int pos = (int)(keys[i] & 0xFFFFFFFFu);
        float mval = (i < T_ - 1) ? asn[i] : 0.f;
        mask0[((size_t)n * T_ + pos) * NC + c] = mval;
    }
}

// ---------------- mask refine + masked logit sums ----------------
__global__ __launch_bounds__(256) void k_refine(const float* __restrict__ seg,
    const float* __restrict__ mask0, const float* __restrict__ logits,
    float* __restrict__ segmask_out,
    float* __restrict__ AN, float* __restrict__ AL, float* __restrict__ BL) {
    __shared__ float rA[256], rB[256], rC[256];
    const int r = blockIdx.x;
    const int n = r / NC, c = r % NC;
    const int tid = threadIdx.x;
    float tol = 0.f, wsum = 0.f;
    for (int t = tid; t < T_; t += 256) {
        float m = mask0[((size_t)n * T_ + t) * NC + c];
        float s = seg[((size_t)n * T_ + t) * NC + c];
        tol += m; wsum += m * s;
    }
    rA[tid] = tol; rB[tid] = wsum;
    __syncthreads();
    for (int st = 128; st > 0; st >>= 1) {
        if (tid < st) { rA[tid] += rA[tid + st]; rB[tid] += rB[tid + st]; }
        __syncthreads();
    }
    float tolT = rA[0];
    float act = rB[0] / ((tolT == 0.f) ? 1.f : tolT);
    __syncthreads();
    float an = 0.f, al = 0.f, bl = 0.f;
    for (int t = tid; t < T_; t += 256) {
        size_t base = (size_t)n * T_ + t;
        float m = mask0[base * NC + c];
        float s = seg[base * NC + c];
        float ref = (s >= act) ? m : 0.f;
        segmask_out[base * NC + c] = ref;
        float lg = logits[base * 21 + c];
        an += ref; al += lg * ref; bl += lg * (1.f - ref);
    }
    rA[tid] = an; rB[tid] = al; rC[tid] = bl;
    __syncthreads();
    for (int st = 128; st > 0; st >>= 1) {
        if (tid < st) { rA[tid] += rA[tid + st]; rB[tid] += rB[tid + st]; rC[tid] += rC[tid + st]; }
        __syncthreads();
    }
    if (tid == 0) { AN[r] = rA[0]; AL[r] = rB[0]; BL[r] = rC[0]; }
}

// ---------------- final act/bkg class softmax ----------------
__global__ void k_final(const float* AN, const float* AL, const float* BL,
                        float* act_out, float* bkg_out) {
    __shared__ float av[NC], bv[NC];
    int n = blockIdx.x;
    int c = threadIdx.x;
    if (c < NC) {
        float an = AN[n * NC + c];
        float bn = (float)T_ - an;
        float and_ = (an == 0.f) ? 1.f : an;
        float bnd  = (bn == 0.f) ? 1.f : bn;
        av[c] = AL[n * NC + c] / and_;
        bv[c] = BL[n * NC + c] / bnd;
    }
    __syncthreads();
    if (c < NC) {
        float mx = -3.4e38f, sm = 0.f;
        for (int i = 0; i < NC; ++i) mx = fmaxf(mx, av[i]);
        for (int i = 0; i < NC; ++i) sm += __expf(av[i] - mx);
        act_out[n * NC + c] = __expf(av[c] - mx) / sm;
        float mx2 = -3.4e38f, sm2 = 0.f;
        for (int i = 0; i < NC; ++i) mx2 = fmaxf(mx2, bv[i]);
        for (int i = 0; i < NC; ++i) sm2 += __expf(bv[i] - mx2);
        bkg_out[n * NC + c] = __expf(bv[c] - mx2) / sm2;
    }
}

extern "C" void kernel_launch(void* const* d_in, const int* in_sizes, int n_in,
                              void* d_out, int out_size, void* d_ws, size_t ws_size,
                              hipStream_t stream) {
    (void)in_sizes; (void)n_in; (void)out_size; (void)ws_size;
    const float* x    = (const float*)d_in[0];
    const float* crw1 = (const float*)d_in[1];  const float* crb1 = (const float*)d_in[2];
    const float* crw2 = (const float*)d_in[3];  const float* crb2 = (const float*)d_in[4];
    const float* cfw1 = (const float*)d_in[5];  const float* cfb1 = (const float*)d_in[6];
    const float* cfw2 = (const float*)d_in[7];  const float* cfb2 = (const float*)d_in[8];
    const float* arw1 = (const float*)d_in[9];  const float* arb1 = (const float*)d_in[10];
    const float* arw2 = (const float*)d_in[11]; const float* arb2 = (const float*)d_in[12];
    const float* afw1 = (const float*)d_in[13]; const float* afb1 = (const float*)d_in[14];
    const float* afw2 = (const float*)d_in[15]; const float* afb2 = (const float*)d_in[16];

    char* ws = (char*)d_ws;
    size_t off = 0;
    auto carve = [&](size_t bytes) -> char* {
        char* p = ws + off;
        off = (off + bytes + 255) & ~(size_t)255;
        return p;
    };
    _Float16* Xp   = (_Float16*)carve((size_t)2 * NB * TP2 * D_ * 2);
    _Float16* Wt   = (_Float16*)carve((size_t)4 * 3 * D_ * D_ * 2);
    _Float16* W2A  = (_Float16*)carve((size_t)32 * 4096 * 2);
    _Float16* H    = (_Float16*)carve((size_t)NB * T_ * 4096 * 2);
    float*    logits = (float*)carve((size_t)NB * T_ * 21 * 4);
    float*    mask0  = (float*)carve((size_t)NB * T_ * NC * 4);
    float*    AN     = (float*)carve(NB * NC * 4);
    float*    AL     = (float*)carve(NB * NC * 4);
    float*    BL     = (float*)carve(NB * NC * 4);

    float* out     = (float*)d_out;
    float* out_act = out;                       // 640
    float* out_bkg = out + 640;                 // 640
    float* out_aas = out + 1280;                // 32768
    float* out_seg = out + 34048;               // 655360
    float* out_msk = out + 689408;              // 655360

    long totalX = (long)2 * NB * TP2 * D_;
    k_prep_x <<<(int)((totalX + 255) / 256), 256, 0, stream>>>(x, Xp, totalX);
    k_prep_w <<<(4 * 3 * D_ * D_) / 256, 256, 0, stream>>>(crw1, cfw1, arw1, afw1, Wt);
    k_prep_w2<<<(32 * 4096) / 256, 256, 0, stream>>>(crw2, cfw2, arw2, afw2, W2A);
    k_conv_gemm<<<dim3(64, NB, 4), 256, 0, stream>>>(Wt, Xp, crb1, cfb1, arb1, afb1, H);
    k_l2_gemm  <<<dim3(8, NB), 256, 0, stream>>>(W2A, H, crb2, cfb2, arb2, afb2, logits);
    k_softmax  <<<(NB * T_) / 256, 256, 0, stream>>>(logits, out_aas, out_seg);
    k_cluster  <<<NB * NC, 512, 0, stream>>>(out_seg, mask0);
    k_refine   <<<NB * NC, 256, 0, stream>>>(out_seg, mask0, logits, out_msk, AN, AL, BL);
    k_final    <<<NB, 32, 0, stream>>>(AN, AL, BL, out_act, out_bkg);
}
